// FP4PerChannelMLP_6975026888712
// MI455X (gfx1250) — matrix operand order repaired and checked by reading, not compiled, not run on hardware
//
#include <hip/hip_runtime.h>
#include <hip/hip_bf16.h>
#include <stdint.h>

// ---------------------------------------------------------------------------
// MI455X fused MLP: y = (gelu((x@w1'+b1)*s1) @ w2' + b2)*s2
// Compute-bound (550 GFLOP vs ~0.4GB traffic) -> bf16 WMMA w/ f32 accumulate.
// Global->LDS staging via gfx1250 async-to-LDS (ASYNCcnt) when available.
// ---------------------------------------------------------------------------

typedef __attribute__((ext_vector_type(16))) __bf16       v16bf;
typedef __attribute__((ext_vector_type(8)))  float        v8f;
typedef __attribute__((ext_vector_type(4)))  unsigned int v4u;
typedef __attribute__((ext_vector_type(2)))  unsigned int v2u;
typedef __attribute__((ext_vector_type(4)))  float        v4f;
typedef int v4i __attribute__((vector_size(4 * sizeof(int))));  // builtin's elem type

#define AS1 __attribute__((address_space(1)))
#define AS3 __attribute__((address_space(3)))

#define TOKENS 16384
#define HID    2048

#define BM 128
#define BN 128
#define BK 32
#define LDT 40          // LDS row stride (elements): BK+8 pad, 80B (16B aligned, 5⊥16 banks)
#define THREADS 256

#if defined(__HIP_DEVICE_COMPILE__) && \
    __has_builtin(__builtin_amdgcn_global_load_async_to_lds_b128) && \
    __has_builtin(__builtin_amdgcn_s_wait_asynccnt)
#define USE_ASYNC_LDS 1
#else
#define USE_ASYNC_LDS 0
#endif

__device__ __forceinline__ uint16_t f32_to_bf16(float f) {
  uint32_t u = __float_as_uint(f);
  uint32_t r = u + 0x7fffu + ((u >> 16) & 1u);   // round-to-nearest-even
  return (uint16_t)(r >> 16);
}

// Bandwidth-bound one-shot f32 -> bf16 downconvert (b128 loads, b64 stores).
__global__ __launch_bounds__(THREADS)
void cvt_f32_to_bf16_kernel(const float* __restrict__ src,
                            uint16_t* __restrict__ dst, int n /*mult of 4*/) {
  int n4 = n >> 2;
  for (int i = blockIdx.x * blockDim.x + threadIdx.x; i < n4;
       i += gridDim.x * blockDim.x) {
    v4f f = ((const v4f*)src)[i];
    v2u o;
    o.x = (uint32_t)f32_to_bf16(f.x) | ((uint32_t)f32_to_bf16(f.y) << 16);
    o.y = (uint32_t)f32_to_bf16(f.z) | ((uint32_t)f32_to_bf16(f.w) << 16);
    ((v2u*)dst)[i] = o;
  }
}

union Frag {          // 32B: matches one 16x32 bf16 WMMA operand per lane
  v4u   q[2];
  v16bf v;
};

// C[M,N] = A[M,K] @ W[N,K]^T, epilogue: t=(acc+bias[n])*scale[n]; optional exact
// GELU; Out is bf16 (FUSE_GELU) or f32. All dims multiples of tile sizes.
template <bool FUSE_GELU>
__global__ __launch_bounds__(THREADS)
void mlp_gemm_wmma(const uint16_t* __restrict__ A,
                   const uint16_t* __restrict__ W,
                   const float* __restrict__ bias,
                   const float* __restrict__ scale,
                   void* __restrict__ Out, int M, int N, int K) {
  __shared__ __align__(16) uint16_t sA[2][BM * LDT];
  __shared__ __align__(16) uint16_t sB[2][BN * LDT];

  const int tid  = threadIdx.x;
  const int lane = tid & 31;
  const int wave = tid >> 5;          // 8 waves (wave32)
  const int wm   = wave & 3;          // 4 waves along M -> 32 rows each
  const int wn   = wave >> 2;         // 2 waves along N -> 64 cols each
  const int m0   = blockIdx.y * BM;
  const int n0   = blockIdx.x * BN;

  // ---- global->LDS staging: 512 16B chunks/tile, 2 per thread ----
  const int crow = tid >> 2;          // rows 0..63 (+64 for second chunk)
  const int ckof = (tid & 3) * 8;     // K offset in elements
  const uint16_t* gA0 = A + (size_t)(m0 + crow) * K + ckof;
  const uint16_t* gA1 = A + (size_t)(m0 + crow + 64) * K + ckof;
  const uint16_t* gW0 = W + (size_t)(n0 + crow) * K + ckof;
  const uint16_t* gW1 = W + (size_t)(n0 + crow + 64) * K + ckof;
  const int lA0 = crow * LDT + ckof, lA1 = (crow + 64) * LDT + ckof;

  v4u rA0, rA1, rB0, rB1;  // sync-fallback staging regs

  auto issueTile = [&](int kt, int buf) {
    const uint16_t* pA0 = gA0 + (size_t)kt * BK;
    const uint16_t* pA1 = gA1 + (size_t)kt * BK;
    const uint16_t* pW0 = gW0 + (size_t)kt * BK;
    const uint16_t* pW1 = gW1 + (size_t)kt * BK;
#if USE_ASYNC_LDS
    __builtin_amdgcn_global_load_async_to_lds_b128(
        (AS1 v4i*)(uintptr_t)pA0, (AS3 v4i*)(uintptr_t)&sA[buf][lA0], 0, 0);
    __builtin_amdgcn_global_load_async_to_lds_b128(
        (AS1 v4i*)(uintptr_t)pA1, (AS3 v4i*)(uintptr_t)&sA[buf][lA1], 0, 0);
    __builtin_amdgcn_global_load_async_to_lds_b128(
        (AS1 v4i*)(uintptr_t)pW0, (AS3 v4i*)(uintptr_t)&sB[buf][lA0], 0, 0);
    __builtin_amdgcn_global_load_async_to_lds_b128(
        (AS1 v4i*)(uintptr_t)pW1, (AS3 v4i*)(uintptr_t)&sB[buf][lA1], 0, 0);
#else
    rA0 = *(const v4u*)pA0;  rA1 = *(const v4u*)pA1;
    rB0 = *(const v4u*)pW0;  rB1 = *(const v4u*)pW1;
    (void)buf;
#endif
  };
  auto finishTile = [&](int buf) {
#if USE_ASYNC_LDS
    __builtin_amdgcn_s_wait_asynccnt(0);
    (void)buf;
#else
    *(v4u*)&sA[buf][lA0] = rA0;  *(v4u*)&sA[buf][lA1] = rA1;
    *(v4u*)&sB[buf][lA0] = rB0;  *(v4u*)&sB[buf][lA1] = rB1;
#endif
  };

  // ---- WMMA fragment LDS offsets (ISA 16-bit A / B per-lane layouts) ----
  const int laneRow = lane & 15;
  const int laneHi  = lane >> 4;
  int aBase[2], bBase[4];
#pragma unroll
  for (int mt = 0; mt < 2; ++mt)      // A lane: K[base..base+7], K[base+16..+23]
    aBase[mt] = (wm * 32 + mt * 16 + laneRow) * LDT + laneHi * 8;
#pragma unroll
  for (int nt = 0; nt < 4; ++nt)      // B lane: K[base..base+7], K[base+16..+23]
    bBase[nt] = (wn * 64 + nt * 16 + laneRow) * LDT + laneHi * 8;

  v8f acc[2][4] = {};

  const int KT = K / BK;
  issueTile(0, 0);
  finishTile(0);
  __syncthreads();

  for (int kt = 0; kt < KT; ++kt) {
    const int buf = kt & 1;
    if (kt + 1 < KT) issueTile(kt + 1, buf ^ 1);

    Frag a[2], b[4];
#pragma unroll
    for (int mt = 0; mt < 2; ++mt) {
      a[mt].q[0] = *(const v4u*)&sA[buf][aBase[mt]];
      a[mt].q[1] = *(const v4u*)&sA[buf][aBase[mt] + 16];
    }
#pragma unroll
    for (int nt = 0; nt < 4; ++nt) {
      b[nt].q[0] = *(const v4u*)&sB[buf][bBase[nt]];
      b[nt].q[1] = *(const v4u*)&sB[buf][bBase[nt] + 16];
    }
#pragma unroll
    for (int mt = 0; mt < 2; ++mt)
#pragma unroll
      for (int nt = 0; nt < 4; ++nt)
        acc[mt][nt] = __builtin_amdgcn_wmma_f32_16x16x32_bf16(
            false, a[mt].v, false, b[nt].v, (short)0, acc[mt][nt], false, false);

    if (kt + 1 < KT) finishTile(buf ^ 1);
    __syncthreads();
  }

  // ---- epilogue: per-channel bias+scale (+ exact GELU), C-layout stores ----
  const int rowBase = m0 + wm * 32 + laneHi * 8;
  const int colBase = n0 + wn * 64 + laneRow;
#pragma unroll
  for (int nt = 0; nt < 4; ++nt) {
    const int col = colBase + nt * 16;
    const float bb = bias[col];
    const float ss = scale[col];
#pragma unroll
    for (int mt = 0; mt < 2; ++mt) {
#pragma unroll
      for (int i = 0; i < 8; ++i) {
        const int row = rowBase + mt * 16 + i;
        float t = (acc[mt][nt][i] + bb) * ss;
        if constexpr (FUSE_GELU) {
          t = 0.5f * t * (1.0f + erff(t * 0.70710678118654752f));
          ((uint16_t*)Out)[(size_t)row * N + col] = f32_to_bf16(t);
        } else {
          ((float*)Out)[(size_t)row * N + col] = t;
        }
      }
    }
  }
}

extern "C" void kernel_launch(void* const* d_in, const int* in_sizes, int n_in,
                              void* d_out, int out_size, void* d_ws, size_t ws_size,
                              hipStream_t stream) {
  (void)in_sizes; (void)n_in; (void)out_size; (void)ws_size;
  const float* x  = (const float*)d_in[0];   // [T, H]
  const float* w1 = (const float*)d_in[1];   // [2H, H]
  const float* b1 = (const float*)d_in[2];   // [2H]
  const float* s1 = (const float*)d_in[3];   // [2H]
  const float* w2 = (const float*)d_in[4];   // [H, 2H]
  const float* b2 = (const float*)d_in[5];   // [H]
  const float* s2 = (const float*)d_in[6];   // [H]
  float* y = (float*)d_out;                  // [T, H] f32

  const int T = TOKENS, H = HID;
  const size_t nx  = (size_t)T * H;          // 33.5M
  const size_t nw  = (size_t)(2 * H) * H;    // 8.4M (w1 and w2)

  uint16_t* xb  = (uint16_t*)d_ws;           // bf16 x
  uint16_t* w1b = xb + nx;                   // bf16 w1
  uint16_t* w2b = w1b + nw;                  // bf16 w2
  uint16_t* hb  = w2b + nw;                  // bf16 gelu output (intermediate)
  // total ws use: (nx + 2*nw + T*2H)*2 = ~235 MB

  cvt_f32_to_bf16_kernel<<<2048, THREADS, 0, stream>>>(x,  xb,  (int)nx);
  cvt_f32_to_bf16_kernel<<<1024, THREADS, 0, stream>>>(w1, w1b, (int)nw);
  cvt_f32_to_bf16_kernel<<<1024, THREADS, 0, stream>>>(w2, w2b, (int)nw);

  dim3 g1((2 * H) / BN, T / BM);   // 32 x 128
  mlp_gemm_wmma<true><<<g1, THREADS, 0, stream>>>(xb, w1b, b1, s1, (void*)hb,
                                                  T, 2 * H, H);
  dim3 g2(H / BN, T / BM);         // 16 x 128
  mlp_gemm_wmma<false><<<g2, THREADS, 0, stream>>>(hb, w2b, b2, s2, (void*)y,
                                                   T, H, 2 * H);
}
